// Attention_33363305955450
// MI455X (gfx1250) — matrix operand order repaired and checked by reading, not compiled, not run on hardware
//
#include <hip/hip_runtime.h>
#include <hip/hip_bf16.h>

// ---------------------------------------------------------------------------
// Attention (B=8, L=4096, E=1024, H=64), flash-attention style, bf16 WMMA,
// with Tensor Data Mover (TDM) staging of the K tile into LDS.
// ---------------------------------------------------------------------------

typedef __attribute__((ext_vector_type(16))) __bf16 v16bf;
typedef __attribute__((ext_vector_type(8)))  float  v8f;
typedef __attribute__((ext_vector_type(4)))  unsigned int u32x4;
typedef __attribute__((ext_vector_type(8)))  int  i32x8;
typedef __attribute__((ext_vector_type(4)))  int  i32x4;

union FragBF {
    unsigned int u[8];
    v16bf v;
};

__device__ __forceinline__ unsigned short f2bf(float f) {
    unsigned int u = __float_as_uint(f);
    unsigned int r = u + 0x7FFFu + ((u >> 16) & 1u);   // round-to-nearest-even
    return (unsigned short)(r >> 16);
}

__device__ __forceinline__ v8f wmma_bf16(const FragBF& a, const FragBF& b, v8f c) {
    return __builtin_amdgcn_wmma_f32_16x16x32_bf16(
        /*neg_a=*/false, a.v, /*neg_b=*/false, b.v,
        /*c_mod=*/(short)0, c, /*reuse_a=*/false, /*reuse_b=*/false);
}

#define EMBED 1024
#define HID   64
#define SEQ   4096
#define BATCH 8

// ---------------------------------------------------------------------------
// Kernel 1: projection  Out[r,h] = sum_e X[r,e] * W[h,e]   (bf16 out, f32 acc)
// Block: 256 threads (8 waves). Each block: 64 rows x 64 cols, K streamed 32.
// ---------------------------------------------------------------------------
#define LSTR 36  // LDS row stride (32 + 4 pad), bf16 elems

__global__ __launch_bounds__(256) void proj_kernel(
    const float* __restrict__ X,       // [nrows, 1024]
    const float* __restrict__ W,       // [64, 1024]
    unsigned short* __restrict__ Out,  // [nrows, 64] bf16
    float scale)
{
    __shared__ unsigned short Xc[64 * LSTR];
    __shared__ unsigned short Wc[64 * LSTR];

    const int tid  = threadIdx.x;
    const int lane = tid & 31;
    const int wave = tid >> 5;
    const int row0 = blockIdx.x * 64;

    // wave -> row strip (wave>>1)*16, col pair (wave&1)*32
    const int m0 = (wave >> 1) * 16;
    const int n0 = (wave & 1) * 32;

    v8f acc0 = {0.f,0.f,0.f,0.f,0.f,0.f,0.f,0.f};
    v8f acc1 = {0.f,0.f,0.f,0.f,0.f,0.f,0.f,0.f};

    const int am   = m0 + (lane & 15);       // A-frag row
    const int kofs = (lane >> 4) * 8;        // A-frag K group offset
    const int bn   = lane & 15;              // B-frag column
    const int bk16 = (lane >> 4) * 16;       // B-frag K half offset

    for (int e0 = 0; e0 < EMBED; e0 += 32) {
        // cooperative load + f32->bf16 convert: 64x32 X chunk, 64x32 W chunk
        {
            const int r = tid >> 2;          // 0..63
            const int c = (tid & 3) * 8;     // 0,8,16,24
            const float* xs = X + (size_t)(row0 + r) * EMBED + e0 + c;
            const float* ws = W + (size_t)r * EMBED + e0 + c;
#pragma unroll
            for (int j = 0; j < 8; ++j) {
                Xc[r * LSTR + c + j] = f2bf(xs[j]);
                Wc[r * LSTR + c + j] = f2bf(ws[j]);
            }
        }
        __syncthreads();

        // A fragment: 16x32 rows m0..m0+15 (contiguous k = 16*(lane/16) + s, same as B)
        FragBF a, b0, b1;
#pragma unroll
        for (int v = 0; v < 8; ++v) {
            int k = bk16 + 2 * v;
            a.u[v] = *(const unsigned int*)&Xc[am * LSTR + k];
        }
        // B fragments: B[e,h] = W[h,e] -> read Wc[h][e]
#pragma unroll
        for (int v = 0; v < 8; ++v) {
            int k = bk16 + 2 * v;
            b0.u[v] = *(const unsigned int*)&Wc[(n0 + bn) * LSTR + k];
            b1.u[v] = *(const unsigned int*)&Wc[(n0 + 16 + bn) * LSTR + k];
        }
        acc0 = wmma_bf16(a, b0, acc0);
        acc1 = wmma_bf16(a, b1, acc1);
        __syncthreads();
    }

    // C/D layout: lane l, vgpr v -> M=(l>=16?8:0)+v, N=l&15
    const int mv = (lane >> 4) * 8;
#pragma unroll
    for (int v = 0; v < 8; ++v) {
        int row = row0 + m0 + mv + v;
        Out[(size_t)row * HID + n0 + bn]      = f2bf(acc0[v] * scale);
        Out[(size_t)row * HID + n0 + 16 + bn] = f2bf(acc1[v] * scale);
    }
}

// ---------------------------------------------------------------------------
// Kernel 2: flash attention over projected Q,K,V (bf16, Q pre-scaled).
// Block: 128 threads (4 waves); block owns (batch b, 64 query rows).
// Wave w owns 16-row strip; loop over key blocks of 32.
// K tile is staged into LDS by the Tensor Data Mover (TENSOR_LOAD_TO_LDS).
// ---------------------------------------------------------------------------
#define LK 64  // K tile LDS stride (row-major, TDM writes it densely)
#define LV 36  // V^T tile LDS stride (32 + 4 pad)
#define LP 36  // P staging stride (32 + 4 pad)

__global__ __launch_bounds__(128) void attn_kernel(
    const unsigned short* __restrict__ Qw,   // [B*L*64] bf16, pre-scaled 1/8
    const unsigned short* __restrict__ Kw,   // [B*L*64] bf16
    const unsigned short* __restrict__ Vw,   // [B*L*64] bf16
    const unsigned char*  __restrict__ mask, // [B*L*L] bool (1 = masked out)
    float* __restrict__ Out)                 // [B*L*64] f32
{
    __shared__ unsigned short Klds[32 * LK];      // [key][h] row-major (TDM)
    __shared__ unsigned short Vlds[64 * LV];      // [h][key] transposed
    __shared__ unsigned short Plds[4 * 16 * LP];  // per-wave P staging

    const int tid  = threadIdx.x;
    const int lane = tid & 31;
    const int wave = tid >> 5;                    // 0..3 -> query strip
    const int b    = blockIdx.y;
    const int q0   = blockIdx.x * 64;

    const int kofs = (lane >> 4) * 8;             // A-frag K group offset
    const int bn   = lane & 15;                   // B-frag col / C-frag col
    const int bk16 = (lane >> 4) * 16;            // B-frag K half offset
    const int mv0  = (lane >> 4) * 8;             // C-frag row offset

    // ---- load Q A-fragments once (h 0..31 and 32..63) ----
    FragBF qa0, qa1;
    {
        const int qrow = q0 + wave * 16 + (lane & 15);
        const unsigned short* qp = Qw + (size_t)(b * SEQ + qrow) * HID;
#pragma unroll
        for (int v = 0; v < 8; ++v) {
            int k = bk16 + 2 * v;
            qa0.u[v] = *(const unsigned int*)(qp + k);
            qa1.u[v] = *(const unsigned int*)(qp + 32 + k);
        }
    }

    v8f o0 = {0.f,0.f,0.f,0.f,0.f,0.f,0.f,0.f};
    v8f o1 = o0, o2 = o0, o3 = o0;
    float psum[8], mrow[8];
#pragma unroll
    for (int v = 0; v < 8; ++v) { psum[v] = 0.f; mrow[v] = -1e30f; }

    const size_t kvbase = (size_t)b * SEQ * HID;
    const unsigned char* mbase =
        mask + (size_t)b * SEQ * SEQ + (size_t)(q0 + wave * 16 + mv0) * SEQ;

    // TDM descriptor constants (2-D tile: 32 rows x 64 bf16, dense)
    const unsigned int kldsAddr = (unsigned int)(size_t)&Klds[0];
    // D# group1: data_size=1(2B), tensor_dim0=64, tensor_dim1=32,
    //            tile_dim0=64, tile_dim1=32, tensor_dim0_stride=64
    const i32x8 tdmG1 = { 0x00010000, (64 << 16), (32 << 16), (64 << 16),
                          32, 64, 0, 0 };
    const i32x4 tdmZ4 = { 0, 0, 0, 0 };           // groups 2/3 unused (2-D)
    const i32x8 tdmZ8 = { 0, 0, 0, 0, 0, 0, 0, 0 };

    for (int k0 = 0; k0 < SEQ; k0 += 32) {
        __syncthreads();  // previous tiles fully consumed
        // ---- TDM: DMA the K tile [32][64] bf16 into LDS (one wave issues)
        if (wave == 0) {
            unsigned long long ga =
                (unsigned long long)(size_t)(Kw + kvbase + (size_t)k0 * HID);
            // D# group0: count=1, lds_addr, global_addr[56:0], type=2
            u32x4 g0 = { 1u, kldsAddr, (unsigned int)ga,
                         (unsigned int)((ga >> 32) & 0x1FFFFFFu) | 0x80000000u };
            __builtin_amdgcn_tensor_load_to_lds(g0, tdmG1, tdmZ4, tdmZ4,
                                                tdmZ8, 0);
        }
        // ---- cooperative stage: V transposed [h][key] (manual, TDM can't
        //      transpose; transposed layout gives contiguous K-pairs for B)
        {
            const int r = tid >> 2;           // key row 0..31 (4 thr/row)
            const int c = (tid & 3) * 16;     // 16 elems each
            const unsigned short* vp = Vw + kvbase + (size_t)(k0 + r) * HID + c;
#pragma unroll
            for (int j = 0; j < 16; ++j)
                Vlds[(c + j) * LV + r] = vp[j];
            if (k0 + 32 < SEQ)                // prefetch next V tile
                __builtin_prefetch(vp + 32 * HID, 0, 1);
        }
        if (wave == 0)
            __builtin_amdgcn_s_wait_tensorcnt(0);   // K tile landed in LDS
        __syncthreads();

        // ---- S = Q . K^T : two 16-key sub-tiles, K-dim = 64 (2 WMMAs each)
        FragBF kb;
        v8f s0 = {0.f,0.f,0.f,0.f,0.f,0.f,0.f,0.f};
        v8f s1 = s0;
#pragma unroll
        for (int v = 0; v < 8; ++v)
            kb.u[v] = *(const unsigned int*)&Klds[bn * LK + bk16 + 2 * v];
        s0 = wmma_bf16(qa0, kb, s0);
#pragma unroll
        for (int v = 0; v < 8; ++v)
            kb.u[v] = *(const unsigned int*)&Klds[bn * LK + 32 + bk16 + 2 * v];
        s0 = wmma_bf16(qa1, kb, s0);
#pragma unroll
        for (int v = 0; v < 8; ++v)
            kb.u[v] = *(const unsigned int*)&Klds[(16 + bn) * LK + bk16 + 2 * v];
        s1 = wmma_bf16(qa0, kb, s1);
#pragma unroll
        for (int v = 0; v < 8; ++v)
            kb.u[v] = *(const unsigned int*)&Klds[(16 + bn) * LK + 32 + bk16 + 2 * v];
        s1 = wmma_bf16(qa1, kb, s1);

        // ---- mask + online softmax (per 16-lane group == one query row set)
        unsigned short* pl = Plds + wave * 16 * LP;
#pragma unroll
        for (int v = 0; v < 8; ++v) {
            const unsigned char* mp = mbase + (size_t)v * SEQ + k0;
            float a = s0[v];
            float c = s1[v];
            if (mp[bn])      a = -1e30f;
            if (mp[16 + bn]) c = -1e30f;
            float mloc = fmaxf(a, c);
            mloc = fmaxf(mloc, __shfl_xor(mloc, 1));
            mloc = fmaxf(mloc, __shfl_xor(mloc, 2));
            mloc = fmaxf(mloc, __shfl_xor(mloc, 4));
            mloc = fmaxf(mloc, __shfl_xor(mloc, 8));
            float mn = fmaxf(mrow[v], mloc);
            float sc = __expf(mrow[v] - mn);
            mrow[v] = mn;
            float e0 = __expf(a - mn);
            float e1 = __expf(c - mn);
            psum[v] = psum[v] * sc + e0 + e1;   // per-lane partial row sum
            o0[v] *= sc; o1[v] *= sc; o2[v] *= sc; o3[v] *= sc;
            // stage P (C layout -> memory [m][k]) for A-frag reload
            pl[(mv0 + v) * LP + bn]      = f2bf(e0);
            pl[(mv0 + v) * LP + 16 + bn] = f2bf(e1);
        }

        // ---- reload P as 16x32 A fragment (same-wave LDS, in-order)
        FragBF pa;
#pragma unroll
        for (int v = 0; v < 8; ++v) {
            int k = bk16 + 2 * v;
            pa.u[v] = *(const unsigned int*)&pl[(lane & 15) * LP + k];
        }

        // ---- O += P . V : V^T LDS gives contiguous K-pairs per column h
        FragBF vb;
#pragma unroll
        for (int v = 0; v < 8; ++v)
            vb.u[v] = *(const unsigned int*)&Vlds[bn * LV + bk16 + 2 * v];
        o0 = wmma_bf16(pa, vb, o0);
#pragma unroll
        for (int v = 0; v < 8; ++v)
            vb.u[v] = *(const unsigned int*)&Vlds[(16 + bn) * LV + bk16 + 2 * v];
        o1 = wmma_bf16(pa, vb, o1);
#pragma unroll
        for (int v = 0; v < 8; ++v)
            vb.u[v] = *(const unsigned int*)&Vlds[(32 + bn) * LV + bk16 + 2 * v];
        o2 = wmma_bf16(pa, vb, o2);
#pragma unroll
        for (int v = 0; v < 8; ++v)
            vb.u[v] = *(const unsigned int*)&Vlds[(48 + bn) * LV + bk16 + 2 * v];
        o3 = wmma_bf16(pa, vb, o3);
    }

    // ---- finalize: reduce row sums across 16 lanes, normalize, store f32
#pragma unroll
    for (int v = 0; v < 8; ++v) {
        float s = psum[v];
        s += __shfl_xor(s, 1);
        s += __shfl_xor(s, 2);
        s += __shfl_xor(s, 4);
        s += __shfl_xor(s, 8);
        float inv = 1.0f / s;
        int row = q0 + wave * 16 + mv0 + v;
        float* op = Out + (size_t)(b * SEQ + row) * HID + bn;
        op[0]  = o0[v] * inv;
        op[16] = o1[v] * inv;
        op[32] = o2[v] * inv;
        op[48] = o3[v] * inv;
    }
}

// ---------------------------------------------------------------------------
// Launch
// ---------------------------------------------------------------------------
extern "C" void kernel_launch(void* const* d_in, const int* in_sizes, int n_in,
                              void* d_out, int out_size, void* d_ws, size_t ws_size,
                              hipStream_t stream) {
    (void)in_sizes; (void)n_in; (void)out_size; (void)ws_size;
    const float* query = (const float*)d_in[0];
    const float* key   = (const float*)d_in[1];
    const float* value = (const float*)d_in[2];
    const unsigned char* mask = (const unsigned char*)d_in[3];  // numpy bool_
    const float* WQ = (const float*)d_in[4];
    const float* WK = (const float*)d_in[5];
    const float* WV = (const float*)d_in[6];

    // workspace: three bf16 [B*L, 64] projections (12 MB total)
    unsigned short* Qw = (unsigned short*)d_ws;
    unsigned short* Kw = Qw + (size_t)BATCH * SEQ * HID;
    unsigned short* Vw = Kw + (size_t)BATCH * SEQ * HID;

    dim3 g1(BATCH * SEQ / 64);
    proj_kernel<<<g1, 256, 0, stream>>>(query, WQ, Qw, 0.125f);  // 1/sqrt(64)
    proj_kernel<<<g1, 256, 0, stream>>>(key,   WK, Kw, 1.0f);
    proj_kernel<<<g1, 256, 0, stream>>>(value, WV, Vw, 1.0f);

    dim3 g2(SEQ / 64, BATCH);
    attn_kernel<<<g2, 128, 0, stream>>>(Qw, Kw, Vw, mask, (float*)d_out);
}
